// GATE21_79319456022826
// MI455X (gfx1250) — matrix operand order repaired and checked by reading, not compiled, not run on hardware
//
#include <hip/hip_runtime.h>
#include <cstdint>
#include <cstddef>

// ---------------- CDNA5 WMMA types & helpers ----------------
typedef __attribute__((ext_vector_type(16))) __bf16 v16bf;
typedef __attribute__((ext_vector_type(8)))  float  v8f;

// pack two f32 -> 2 x bf16 in one dword (low = x, high = y) with truncation;
// single v_perm_b32: bytes {uy[3],uy[2],ux[3],ux[2]}
__device__ __forceinline__ unsigned pack2bf(float x, float y) {
  return __builtin_amdgcn_perm(__float_as_uint(y), __float_as_uint(x), 0x07060302u);
}

// monotone float <-> uint encoding for atomicMax on floats
__device__ __forceinline__ unsigned ford(float f) {
  unsigned u = __float_as_uint(f);
  return (u & 0x80000000u) ? ~u : (u | 0x80000000u);
}
__device__ __forceinline__ float ordf(unsigned u) {
  return __uint_as_float((u & 0x80000000u) ? (u & 0x7FFFFFFFu) : ~u);
}
#define ORD_NEG_INF 0x007FFFFFu   // ford(-inf)

// Cooperative stage of W[0..K, col0..col0+16) into LDS, bf16 (trunc), transposed:
// Ws[c][k] with row pitch K+2 (odd dword pitch -> no bank conflicts).
template <int K>
__device__ __forceinline__ void stage_w(const float* __restrict__ W, int ldw, int col0,
                                        unsigned short* __restrict__ Ws) {
  const int t = (int)(threadIdx.y * 32 + threadIdx.x);   // 0..127
  constexpr int KP = K + 2;
  for (int i = t; i < K * 16; i += 128) {
    const int k = i >> 4, c = i & 15;
    Ws[c * KP + k] =
        (unsigned short)(__float_as_uint(W[(size_t)k * ldw + (col0 + c)]) >> 16);
  }
}

// One wave computes one 16x16 f32 tile: Out = act(A * W + bias).
// A via af2(k) -> float2 (pair k,k+1 of this lane's row); B from LDS (bf16 pairs).
// K, ACT compile-time => fully unrolled, branch-free inner loop.
template <int K, int ACT, typename AF2>
__device__ __forceinline__ void wmma_compute(AF2 af2,
                                             const unsigned short* __restrict__ Ws,
                                             const float* __restrict__ bias,
                                             float* __restrict__ Out, int ldo,
                                             int row0, int col0)
{
  const int half = (int)((threadIdx.x >> 4) & 1);
  const int l16  = (int)(threadIdx.x & 15);
  constexpr int KP = K + 2;
  const unsigned* __restrict__ Wrow =
      (const unsigned*)(Ws + (size_t)l16 * KP);     // packed bf16 pairs, this column
  v8f acc = {};
#pragma unroll
  for (int k0 = 0; k0 < K; k0 += 32) {
    union Frag { v16bf v; unsigned u[8]; };
    Frag a, b;
#pragma unroll
    for (int v = 0; v < 8; ++v) {
      // A 16x32 bf16 layout: vgpr v holds K = g*16 + half*8 + (v%4)*2 (+1)
      const int ka = k0 + ((v >> 2) << 4) + (half << 3) + ((v & 3) << 1);
      if (ka < K) {                       // folds at compile time (K,half bounded)
        float2 av = af2(ka);
        a.u[v] = pack2bf(av.x, av.y);
      } else {
        a.u[v] = 0u;
      }
      // B 32x16 bf16 layout: lane col = l16, vgpr v holds K = g*16 + half*8 + (v%4)*2 (+1)
      const int kb = k0 + ((v >> 2) << 4) + (half << 3) + ((v & 3) << 1);
      b.u[v] = (kb < K) ? Wrow[kb >> 1] : 0u;
    }
    acc = __builtin_amdgcn_wmma_f32_16x16x32_bf16(false, a.v, false, b.v,
                                                  (short)0, acc, false, false);
  }
  const int coln = col0 + l16;
  const float bv = bias ? bias[coln] : 0.0f;
#pragma unroll
  for (int v = 0; v < 8; ++v) {
    const int m = row0 + v + (half << 3);   // C/D layout: M = v + 8*half
    float val = acc[v] + bv;
    if (ACT) val = fmaxf(val, 0.0f);
    Out[(size_t)m * ldo + coln] = val;
  }
}

// ---------------- WMMA GEMM kernels (4 waves/block, 64 rows/block) ----------------
template <int K, int ACT>
__global__ void __launch_bounds__(128)
k_gemm_t(const float* __restrict__ A, int lda,
         const float* __restrict__ W, int ldw,
         const float* __restrict__ bias,
         float* __restrict__ Out, int ldo, int M)
{
  __shared__ unsigned short Ws[16 * (K + 2)];
  const int col0 = blockIdx.x * 16;
  stage_w<K>(W, ldw, col0, Ws);
  __syncthreads();
  const int row0 = (blockIdx.y * 4 + threadIdx.y) * 16;
  if (row0 >= M) return;
  const float* __restrict__ Ar = A + (size_t)(row0 + (threadIdx.x & 15)) * lda;
  __builtin_prefetch(Ar, 0, 3);                       // global_prefetch_b8
  auto af2 = [&](int k) -> float2 { return *(const float2*)(Ar + k); };
  wmma_compute<K, ACT>(af2, Ws, bias, Out, ldo, row0, col0);
}

// Edge MLP layer 1: A-row e = [ x[src[e]] (64) | x[tgt[e]] (64) | ea[e] (ED) ]
template <int K, int ED, int ACT>
__global__ void __launch_bounds__(128)
k_edge_gemm_t(const float* __restrict__ X,
              const int* __restrict__ src, const int* __restrict__ tgt,
              const float* __restrict__ EA,
              const float* __restrict__ W, int ldw,
              const float* __restrict__ bias,
              float* __restrict__ Out, int ldo, int M)
{
  __shared__ unsigned short Ws[16 * (K + 2)];
  const int col0 = blockIdx.x * 16;
  stage_w<K>(W, ldw, col0, Ws);
  __syncthreads();
  const int row0 = (blockIdx.y * 4 + threadIdx.y) * 16;
  if (row0 >= M) return;
  const int e = row0 + (threadIdx.x & 15);
  const float* __restrict__ Xs = X + (size_t)src[e] * 64;
  const float* __restrict__ Xt = X + (size_t)tgt[e] * 64;
  const float* __restrict__ Ae = EA + (size_t)e * ED;
  __builtin_prefetch(Xs, 0, 3);
  __builtin_prefetch(Xt, 0, 3);
  auto af2 = [&](int k) -> float2 {      // segment selects fold at compile time
    if (k < 64)  return *(const float2*)(Xs + k);
    if (k < 128) return *(const float2*)(Xt + (k - 64));
    return *(const float2*)(Ae + (k - 128));
  };
  wmma_compute<K, ACT>(af2, Ws, bias, Out, ldo, row0, col0);
}

// Global-attn MLP layer 1: A-row n = [ xn[n] (64) | u[batch[n]] (384) ]
template <int K, int ACT>
__global__ void __launch_bounds__(128)
k_nodeu_gemm_t(const float* __restrict__ XN,
               const float* __restrict__ U,
               const int* __restrict__ batch,
               const float* __restrict__ W, int ldw,
               const float* __restrict__ bias,
               float* __restrict__ Out, int ldo, int M)
{
  __shared__ unsigned short Ws[16 * (K + 2)];
  const int col0 = blockIdx.x * 16;
  stage_w<K>(W, ldw, col0, Ws);
  __syncthreads();
  const int row0 = (blockIdx.y * 4 + threadIdx.y) * 16;
  if (row0 >= M) return;
  const int n = row0 + (threadIdx.x & 15);
  const float* __restrict__ Xr = XN + (size_t)n * 64;
  const float* __restrict__ Ur = U + (size_t)batch[n] * 384;
  __builtin_prefetch(Ur, 0, 3);
  auto af2 = [&](int k) -> float2 {
    if (k < 64) return *(const float2*)(Xr + k);
    return *(const float2*)(Ur + (k - 64));
  };
  wmma_compute<K, ACT>(af2, Ws, bias, Out, ldo, row0, col0);
}

// ---------------- scalar glue kernels ----------------
__global__ void k_fill_u32(unsigned* __restrict__ p, unsigned v, size_t n) {
  size_t i = blockIdx.x * (size_t)blockDim.x + threadIdx.x;
  if (i < n) p[i] = v;
}

__global__ void k_dot1(const float* __restrict__ A, const float* __restrict__ w,
                       const float* __restrict__ b, float* __restrict__ out,
                       int M, int K) {
  int r = blockIdx.x * blockDim.x + threadIdx.x;
  if (r >= M) return;
  const float4* __restrict__ A4 = (const float4*)(A + (size_t)r * K);
  const float4* __restrict__ w4 = (const float4*)w;
  float acc = b[0];
  for (int k = 0; k < (K >> 2); ++k) {
    float4 av = A4[k], wv = w4[k];
    acc += av.x * wv.x + av.y * wv.y + av.z * wv.z + av.w * wv.w;
  }
  out[r] = acc;
}

__global__ void k_alpha_max(const float* __restrict__ xl, const float* __restrict__ xr,
                            const float* __restrict__ ee,
                            const int* __restrict__ src, const int* __restrict__ tgt,
                            const float* __restrict__ att,
                            float* __restrict__ alpha, unsigned* __restrict__ segmax,
                            int E4) {
  int i = blockIdx.x * blockDim.x + threadIdx.x;
  if (i >= E4) return;
  int e = i >> 2, h = i & 3;
  int s = src[e], t = tgt[e];
  const float4* pl = (const float4*)(xl + (size_t)s * 64 + h * 16);
  const float4* pr = (const float4*)(xr + (size_t)t * 64 + h * 16);
  const float4* pe = (const float4*)(ee + (size_t)e * 64 + h * 16);
  const float4* pa = (const float4*)(att + h * 16);
  float acc = 0.0f;
#pragma unroll
  for (int c = 0; c < 4; ++c) {
    float4 vl = pl[c], vr = pr[c], ve = pe[c], va = pa[c];
    float z0 = vl.x + vr.x + ve.x, z1 = vl.y + vr.y + ve.y;
    float z2 = vl.z + vr.z + ve.z, z3 = vl.w + vr.w + ve.w;
    z0 = (z0 > 0.0f) ? z0 : 0.2f * z0;       // leaky_relu 0.2
    z1 = (z1 > 0.0f) ? z1 : 0.2f * z1;
    z2 = (z2 > 0.0f) ? z2 : 0.2f * z2;
    z3 = (z3 > 0.0f) ? z3 : 0.2f * z3;
    acc += va.x * z0 + va.y * z1 + va.z * z2 + va.w * z3;
  }
  alpha[i] = acc;
  atomicMax(&segmax[t * 4 + h], ford(acc));
}

__global__ void k_alpha_exp(float* __restrict__ alpha, const int* __restrict__ tgt,
                            const unsigned* __restrict__ segmax,
                            float* __restrict__ segsum, int E4) {
  int i = blockIdx.x * blockDim.x + threadIdx.x;
  if (i >= E4) return;
  int e = i >> 2, h = i & 3;
  int t = tgt[e];
  float ex = expf(alpha[i] - ordf(segmax[t * 4 + h]));
  alpha[i] = ex;
  atomicAdd(&segsum[t * 4 + h], ex);
}

__global__ void k_scatter_msg(const float* __restrict__ alpha,
                              const float* __restrict__ segsum,
                              const float* __restrict__ xl,
                              const int* __restrict__ src, const int* __restrict__ tgt,
                              float* __restrict__ xn, int E4) {
  int i = blockIdx.x * blockDim.x + threadIdx.x;
  if (i >= E4) return;
  int e = i >> 2, h = i & 3;
  int s = src[e], t = tgt[e];
  float w = alpha[i] / segsum[t * 4 + h];
  const float4* pl = (const float4*)(xl + (size_t)s * 64 + h * 16);
  float* po = xn + (size_t)t * 64 + h * 16;
#pragma unroll
  for (int c = 0; c < 4; ++c) {
    float4 v = pl[c];
    atomicAdd(&po[4 * c + 0], v.x * w);
    atomicAdd(&po[4 * c + 1], v.y * w);
    atomicAdd(&po[4 * c + 2], v.z * w);
    atomicAdd(&po[4 * c + 3], v.w * w);
  }
}

__global__ void k_bias_relu(float* __restrict__ x, const float* __restrict__ bias,
                            int M, int D) {
  size_t i = blockIdx.x * (size_t)blockDim.x + threadIdx.x;
  if (i >= (size_t)M * D) return;
  int f = (int)(i % D);
  x[i] = fmaxf(x[i] + bias[f], 0.0f);
}

__global__ void k_gmax(const float* __restrict__ s, const int* __restrict__ batch,
                       unsigned* __restrict__ gmax, int N) {
  int n = blockIdx.x * blockDim.x + threadIdx.x;
  if (n >= N) return;
  atomicMax(&gmax[batch[n]], ford(s[n]));
}

__global__ void k_gexp(float* __restrict__ s, const int* __restrict__ batch,
                       const unsigned* __restrict__ gmax, float* __restrict__ gsum,
                       int N) {
  int n = blockIdx.x * blockDim.x + threadIdx.x;
  if (n >= N) return;
  int g = batch[n];
  float ex = expf(s[n] - ordf(gmax[g]));
  s[n] = ex;
  atomicAdd(&gsum[g], ex);
}

__global__ void k_pool(const float* __restrict__ s, const float* __restrict__ gsum,
                       const int* __restrict__ batch, const float* __restrict__ xn,
                       float* __restrict__ pooled, float* __restrict__ aout, int N) {
  int i = blockIdx.x * blockDim.x + threadIdx.x;
  if (i >= N * 64) return;
  int n = i >> 6, c = i & 63;
  int g = batch[n];
  float a = s[n] / gsum[g];
  atomicAdd(&pooled[(size_t)g * 64 + c], xn[(size_t)n * 64 + c] * a);
  if (c == 0) aout[n] = a;
}

__global__ void k_cat2(const float* __restrict__ U, const float* __restrict__ P,
                       float* __restrict__ out, int G) {
  int i = blockIdx.x * blockDim.x + threadIdx.x;
  if (i >= G * 448) return;
  int g = i / 448, j = i % 448;
  out[i] = (j < 384) ? U[(size_t)g * 384 + j] : P[(size_t)g * 64 + (j - 384)];
}

// biased batchnorm stats: st[0..D)=sum, st[D..2D)=sum of squares
__global__ void k_bn_stats(const float* __restrict__ X, int M, int D,
                           float* __restrict__ st) {
  int t = blockIdx.x * blockDim.x + threadIdx.x;   // launched <<<D,256>>>
  int f = t % D, s0 = t / D;
  float sum = 0.0f, sq = 0.0f;
  for (int r = s0; r < M; r += 256) {
    float v = X[(size_t)r * D + f];
    sum += v; sq += v * v;
  }
  atomicAdd(&st[f], sum);
  atomicAdd(&st[D + f], sq);
}

__global__ void k_bn_apply(float* __restrict__ X, int M, int D,
                           const float* __restrict__ st,
                           const float* __restrict__ g, const float* __restrict__ b) {
  size_t i = blockIdx.x * (size_t)blockDim.x + threadIdx.x;
  if (i >= (size_t)M * D) return;
  int f = (int)(i % D);
  float mean = st[f] / (float)M;
  float var  = st[D + f] / (float)M - mean * mean;
  X[i] = (X[i] - mean) * rsqrtf(var + 1e-5f) * g[f] + b[f];
}

// ---------------- host orchestration ----------------
struct LayerP {
  const float *eW1, *eW2, *eb1, *eb2;                       // edge mlp
  const float *We, *Wl, *Wr, *att, *bias, *bl, *br;         // gat
  const float *aW1, *aW2, *ab1, *ab2;                       // glob.attn
  const float *mW1, *mW2, *mb1, *mb2;                       // glob.mlp
};

extern "C" void kernel_launch(void* const* d_in, const int* in_sizes, int n_in,
                              void* d_out_, int out_size, void* d_ws, size_t ws_size,
                              hipStream_t stream)
{
  (void)n_in; (void)out_size; (void)ws_size;
  const float* x0    = (const float*)d_in[0];
  const float* eattr = (const float*)d_in[1];
  const float* uemb  = (const float*)d_in[2];
  const int*   eidx  = (const int*)d_in[3];
  const int*   batch = (const int*)d_in[4];
  const int N = in_sizes[0] / 128;
  const int E = in_sizes[1] / 16;
  const int G = in_sizes[2] / 384;
  const int* src = eidx;        // edge_index[0]
  const int* tgt = eidx + E;    // edge_index[1]

  // params: jax pytree flatten order (dict keys sorted at every level)
#define PF(i) ((const float*)d_in[(i)])
  const float* bn_edge_b = PF(5);  const float* bn_edge_g = PF(6);
  const float* bn_node_b = PF(7);  const float* bn_node_g = PF(8);
  const float* bn_u_b    = PF(9);  const float* bn_u_g    = PF(10);
  const float* fc1W = PF(11); const float* fc1b = PF(12);
  const float* fc2W = PF(13); const float* fc2b = PF(14);
  LayerP L1 = { PF(15), PF(16), PF(17), PF(18),
                PF(19), PF(20), PF(21), PF(22), PF(23), PF(24), PF(25),
                PF(26), PF(27), PF(28), PF(29),
                PF(30), PF(31), PF(32), PF(33) };
  LayerP L2 = { PF(34), PF(35), PF(36), PF(37),
                PF(38), PF(39), PF(40), PF(41), PF(42), PF(43), PF(44),
                PF(45), PF(46), PF(47), PF(48),
                PF(49), PF(50), PF(51), PF(52) };
  const float* ntW1 = PF(53); const float* ntW2 = PF(54);
  const float* ntb1 = PF(55); const float* ntb2 = PF(56);
#undef PF

  // workspace carve-up (floats)
  float* w = (float*)d_ws;
  auto alloc = [&](size_t n) { float* p = w; w += n; return p; };
  float* xbuf   = alloc((size_t)N * 64);
  float* hbuf   = alloc((size_t)N * 256);
  float* xlb    = alloc((size_t)N * 64);
  float* xrb    = alloc((size_t)N * 64);
  float* xnb    = alloc((size_t)N * 64);
  float* nah    = alloc((size_t)N * 64);
  float* scores = alloc((size_t)N);
  float* segmax = alloc((size_t)N * 4);
  float* segsum = alloc((size_t)N * 4);
  float* gmax   = alloc((size_t)G);
  float* gsum   = alloc((size_t)G);
  float* pooled = alloc((size_t)G * 64);
  float* ucat   = alloc((size_t)G * 448);
  float* uh     = alloc((size_t)G * 384);
  float* ubuf   = alloc((size_t)G * 384);
  float* u2b    = alloc((size_t)G * 384);
  float* ghb    = alloc((size_t)G * 64);
  float* stats  = alloc(768);
  float* alpha  = alloc((size_t)E * 4);
  float* eB     = alloc((size_t)E * 64);   // ea2 / ea (bn'ed in place)
  float* eC     = alloc((size_t)E * 64);   // edge hidden, then ee

  float* out = (float*)d_out_;
  float* a1  = out + G;
  float* a2  = out + G + N;

  const dim3 GB(32, 4);   // 4 wave32s / block, one 16x16 tile per wave
#define GEMM(Kc, ACTc, A, lda, Wm, Nc, bias, O, M)                                  \
  k_gemm_t<Kc, ACTc><<<dim3((Nc) / 16, (M) / 64), GB, 0, stream>>>(                 \
      A, lda, Wm, Nc, bias, O, Nc, M)

  auto fillu = [&](void* p, unsigned v, size_t n) {
    k_fill_u32<<<dim3((unsigned)((n + 255) / 256)), dim3(256), 0, stream>>>(
        (unsigned*)p, v, n);
  };
  auto b256 = [](long n) { return dim3((unsigned)((n + 255) / 256)); };

  // edge-MLP layer-1 launchers (compile-time K / edge-dim)
  auto edgeL1 = [&](const float* Xin, const LayerP& L) {
    k_edge_gemm_t<144, 16, 1><<<dim3(4, E / 64), GB, 0, stream>>>(
        Xin, src, tgt, eattr, L.eW1, 64, L.eb1, eC, 64, E);
  };
  auto edgeL2 = [&](const float* Xin, const LayerP& L) {
    k_edge_gemm_t<192, 64, 1><<<dim3(4, E / 64), GB, 0, stream>>>(
        Xin, src, tgt, eB, L.eW1, 64, L.eb1, eC, 64, E);
  };

  auto run_meta = [&](const float* Xin, auto edge1, const float* Uin,
                      const LayerP& L, float* XnOut, float* EaOut, float* Uout,
                      float* Aout) {
    // EdgeModel: fused gather-GEMM (no concat materialization)
    edge1(Xin, L);                                            // -> eC (hidden)
    GEMM(64, 0, eC, 64, L.eW2, 64, L.eb2, EaOut, E);          // ea2
    // GATv2 projections
    GEMM(64, 0, Xin, 64, L.Wl, 64, L.bl, xlb, N);
    GEMM(64, 0, Xin, 64, L.Wr, 64, L.br, xrb, N);
    GEMM(64, 0, EaOut, 64, L.We, 64, (const float*)nullptr, eC, E);   // ee
    fillu(segmax, ORD_NEG_INF, (size_t)N * 4);
    fillu(segsum, 0u, (size_t)N * 4);
    fillu(XnOut, 0u, (size_t)N * 64);
    const int E4 = E * 4;
    k_alpha_max<<<b256(E4), 256, 0, stream>>>(xlb, xrb, eC, src, tgt, L.att,
                                              alpha, (unsigned*)segmax, E4);
    k_alpha_exp<<<b256(E4), 256, 0, stream>>>(alpha, tgt, (unsigned*)segmax,
                                              segsum, E4);
    k_scatter_msg<<<b256(E4), 256, 0, stream>>>(alpha, segsum, xlb, src, tgt,
                                                XnOut, E4);
    k_bias_relu<<<b256((long)N * 64), 256, 0, stream>>>(XnOut, L.bias, N, 64);
    // GlobalModel: attention pooling
    k_nodeu_gemm_t<448, 1><<<dim3(4, N / 64), GB, 0, stream>>>(
        XnOut, Uin, batch, L.aW1, 64, L.ab1, nah, 64, N);
    k_dot1<<<b256(N), 256, 0, stream>>>(nah, L.aW2, L.ab2, scores, N, 64);
    fillu(gmax, ORD_NEG_INF, (size_t)G);
    fillu(gsum, 0u, (size_t)G);
    fillu(pooled, 0u, (size_t)G * 64);
    k_gmax<<<b256(N), 256, 0, stream>>>(scores, batch, (unsigned*)gmax, N);
    k_gexp<<<b256(N), 256, 0, stream>>>(scores, batch, (unsigned*)gmax, gsum, N);
    k_pool<<<b256((long)N * 64), 256, 0, stream>>>(scores, gsum, batch, XnOut,
                                                   pooled, Aout, N);
    k_cat2<<<b256((long)G * 448), 256, 0, stream>>>(Uin, pooled, ucat, G);
    GEMM(448, 1, ucat, 448, L.mW1, 384, L.mb1, uh, G);
    GEMM(384, 0, uh, 384, L.mW2, 384, L.mb2, Uout, G);        // u2
  };

  auto bn = [&](float* X, int M, int D, const float* gg, const float* bb) {
    fillu(stats, 0u, (size_t)2 * D);
    k_bn_stats<<<dim3((unsigned)D), dim3(256), 0, stream>>>(X, M, D, stats);
    k_bn_apply<<<b256((long)M * D), 256, 0, stream>>>(X, M, D, stats, gg, bb);
  };

  // NodeTransform: [N,128] -> relu 256 -> 64
  GEMM(128, 1, x0, 128, ntW1, 256, ntb1, hbuf, N);
  GEMM(256, 0, hbuf, 256, ntW2, 64, ntb2, xbuf, N);

  // MetaLayer 1 (edge_attr dim 16, K=144)
  run_meta(xbuf, edgeL1, uemb, L1, xnb, eB, ubuf, a1);

  // BatchNorms (in place)
  bn(xnb, N, 64, bn_node_g, bn_node_b);
  bn(eB, E, 64, bn_edge_g, bn_edge_b);
  bn(ubuf, G, 384, bn_u_g, bn_u_b);

  // MetaLayer 2 (edge feature dim 64, K=192); xbuf reused as xn2 accumulator,
  // eB safely overwritten after its rows are consumed by the first edge GEMM.
  run_meta(xnb, edgeL2, ubuf, L2, xbuf, eB, u2b, a2);

  // Readout: relu(u2 @ fc1) @ fc2 + b
  GEMM(384, 1, u2b, 384, fc1W, 64, fc1b, ghb, G);
  k_dot1<<<b256(G), 256, 0, stream>>>(ghb, fc2W, fc2b, out, G, 64);
#undef GEMM
}
